// Encoder_28836410425667
// MI455X (gfx1250) — matrix operand order repaired and checked by reading, not compiled, not run on hardware
//
#include <hip/hip_runtime.h>
#include <hip/hip_bf16.h>
#include <math.h>

// Problem dims (fixed by the reference)
#define B_DIM 512
#define T_LEN 256
#define N_DIM 128
#define M_DIM 128

// ---------------------------------------------------------------------------
// Types for WMMA fragments (gfx1250, wave32)
// ---------------------------------------------------------------------------
typedef __attribute__((ext_vector_type(16))) __bf16          v16bf;
typedef __attribute__((ext_vector_type(8)))  float           v8f;
typedef __attribute__((ext_vector_type(8)))  unsigned short  u16x8;

union Frag16 {
    v16bf bf;
    u16x8 h[2];
};

// round-to-nearest-even fp32 -> bf16 (bit pattern)
__device__ __forceinline__ unsigned short f2bf(float x) {
    unsigned int u = __float_as_uint(x);
    u += 0x7FFFu + ((u >> 16) & 1u);
    return (unsigned short)(u >> 16);
}

__device__ __forceinline__ float sigmoidf_(float x) {
    return 1.0f / (1.0f + __expf(-x));
}

// gfx1250 has V_TANH_F32 in the TRANS unit; use the builtin when available.
__device__ __forceinline__ float htanh(float x) {
#if defined(__has_builtin)
#if __has_builtin(__builtin_amdgcn_tanhf)
    return __builtin_amdgcn_tanhf(x);
#else
    return tanhf(x);
#endif
#else
    return tanhf(x);
#endif
}

// ---------------------------------------------------------------------------
// GEMM: C(MxN) = A(MxK,bf16,row-major) * B(KxN,bf16, given as BT = NxK row-major)
//       + bias(N).  f32 accumulate via v_wmma_f32_16x16x32_bf16.
// Each wave computes a 32x32 tile of C as 2x2 16x16 fragments.
// blockDim.x must be 256 (8 waves).
// ---------------------------------------------------------------------------
__global__ __launch_bounds__(256)
void gemm_bf16_wmma(const unsigned short* __restrict__ A,   // M x K
                    const unsigned short* __restrict__ BT,  // N x K
                    const float* __restrict__ bias,         // N
                    float* __restrict__ C,                  // M x N
                    int M, int N, int K)
{
    const int lane = threadIdx.x & 31;
    const int wave = threadIdx.x >> 5;
    const int ntn  = N >> 5;                       // 32-wide col tiles
    const int wid  = blockIdx.x * 8 + wave;
    const int total = (M >> 5) * ntn;
    if (wid >= total) return;                      // wave-uniform: EXEC all-1s inside

    const int tm = (wid / ntn) << 5;               // C row base
    const int tn = (wid % ntn) << 5;               // C col base

    v8f acc00 = {}, acc01 = {}, acc10 = {}, acc11 = {};

    // A-fragment addressing (16-bit A 16x32 layout, ISA 7.12.2)
    const int arow0 = tm + (lane & 15);
    const int arow1 = arow0 + 16;
    const int koffA = (lane < 16) ? 0 : 8;
    // B-fragment addressing from BT (N x K): same K layout as A (8 + 8 split)
    const int bcol0 = tn + (lane & 15);
    const int bcol1 = bcol0 + 16;
    const int koffB = (lane < 16) ? 0 : 8;

    const unsigned short* pa0 = A + (size_t)arow0 * K;
    const unsigned short* pa1 = A + (size_t)arow1 * K;
    const unsigned short* pb0 = BT + (size_t)bcol0 * K + koffB;
    const unsigned short* pb1 = BT + (size_t)bcol1 * K + koffB;

    for (int k = 0; k < K; k += 32) {
        Frag16 a0, a1, b0, b1;
        a0.h[0] = *(const u16x8*)(pa0 + k + koffA);
        a0.h[1] = *(const u16x8*)(pa0 + k + 16 + koffA);
        a1.h[0] = *(const u16x8*)(pa1 + k + koffA);
        a1.h[1] = *(const u16x8*)(pa1 + k + 16 + koffA);
        b0.h[0] = *(const u16x8*)(pb0 + k);
        b0.h[1] = *(const u16x8*)(pb0 + k + 16);
        b1.h[0] = *(const u16x8*)(pb1 + k);
        b1.h[1] = *(const u16x8*)(pb1 + k + 16);

        acc00 = __builtin_amdgcn_wmma_f32_16x16x32_bf16(false, a0.bf, false, b0.bf,
                                                        (short)0, acc00, false, false);
        acc01 = __builtin_amdgcn_wmma_f32_16x16x32_bf16(false, a0.bf, false, b1.bf,
                                                        (short)0, acc01, false, false);
        acc10 = __builtin_amdgcn_wmma_f32_16x16x32_bf16(false, a1.bf, false, b0.bf,
                                                        (short)0, acc10, false, false);
        acc11 = __builtin_amdgcn_wmma_f32_16x16x32_bf16(false, a1.bf, false, b1.bf,
                                                        (short)0, acc11, false, false);
    }

    // C/D layout: lane L holds N = tn + (L&15) (+16), M = tm + v + (L<16?0:8) (+16)
    const int ccol0 = tn + (lane & 15);
    const int ccol1 = ccol0 + 16;
    const int crow  = tm + ((lane < 16) ? 0 : 8);
    const float bias0 = bias ? bias[ccol0] : 0.0f;
    const float bias1 = bias ? bias[ccol1] : 0.0f;
#pragma unroll
    for (int v = 0; v < 8; ++v) {
        const int r0 = crow + v;
        const int r1 = r0 + 16;
        C[(size_t)r0 * N + ccol0] = acc00[v] + bias0;
        C[(size_t)r0 * N + ccol1] = acc01[v] + bias1;
        C[(size_t)r1 * N + ccol0] = acc10[v] + bias0;
        C[(size_t)r1 * N + ccol1] = acc11[v] + bias1;
    }
}

// ---------------------------------------------------------------------------
// Attention kernel: one block per batch element b (256 threads = 8 waves).
//   e[b,n]   = sum_t tanh(hs[b,t] + ux[b,n,t]) * ve[t] + bv
//   attn     = softmax_n(e)
//   az[b,n]  = bf16(attn[n] * X[b,t_step,n])        (x_tilde part of fused A)
// ux is streamed as float4 (global_load_b128); hs/ve read from LDS as float4.
// ---------------------------------------------------------------------------
__global__ __launch_bounds__(256)
void attn_kernel(const float* __restrict__ hs,   // B x T
                 const float* __restrict__ ux,   // B x N x T
                 const float* __restrict__ ve,   // T
                 const float* __restrict__ bv,   // 1
                 const float* __restrict__ X,    // B x T x N
                 unsigned short* __restrict__ az,// B x (N+M), cols [0,N) = x_tilde
                 int t_step)
{
    const int b    = blockIdx.x;
    const int tid  = threadIdx.x;
    const int lane = tid & 31;
    const int wave = tid >> 5;

    __shared__ float4 hs4[T_LEN / 4];
    __shared__ float4 ve4[T_LEN / 4];
    __shared__ float  e_s[N_DIM];
    __shared__ float  red[N_DIM];

    ((float*)hs4)[tid] = hs[(size_t)b * T_LEN + tid];
    ((float*)ve4)[tid] = ve[tid];
    __syncthreads();

    const float* uxb = ux + (size_t)b * N_DIM * T_LEN;
    const float  bvv = bv[0];

    for (int n = wave; n < N_DIM; n += 8) {
        const float4* p4 = (const float4*)(uxb + (size_t)n * T_LEN);
        float acc = 0.0f;
#pragma unroll
        for (int j = 0; j < T_LEN / 4; j += 32) {       // 2 iterations (T=256)
            const int t4 = j + lane;
            const float4 u  = p4[t4];
            const float4 h4 = hs4[t4];
            const float4 v4 = ve4[t4];
            acc += htanh(h4.x + u.x) * v4.x;
            acc += htanh(h4.y + u.y) * v4.y;
            acc += htanh(h4.z + u.z) * v4.z;
            acc += htanh(h4.w + u.w) * v4.w;
        }
#pragma unroll
        for (int off = 16; off > 0; off >>= 1)
            acc += __shfl_xor(acc, off, 32);
        if (lane == 0) e_s[n] = acc + bvv;
    }
    __syncthreads();

    // softmax over n = 0..127 (numerically stable)
    if (tid < N_DIM) red[tid] = e_s[tid];
    __syncthreads();
    for (int s = N_DIM / 2; s > 0; s >>= 1) {
        if (tid < s) red[tid] = fmaxf(red[tid], red[tid + s]);
        __syncthreads();
    }
    const float mx = red[0];
    __syncthreads();
    if (tid < N_DIM) red[tid] = __expf(e_s[tid] - mx);
    __syncthreads();
    for (int s = N_DIM / 2; s > 0; s >>= 1) {
        if (tid < s) red[tid] += red[tid + s];
        __syncthreads();
    }
    const float inv = 1.0f / red[0];

    if (tid < N_DIM) {
        const float a  = __expf(e_s[tid] - mx) * inv;
        const float xt = X[((size_t)b * T_LEN + t_step) * N_DIM + tid];
        az[(size_t)b * (N_DIM + M_DIM) + tid] = f2bf(a * xt);
    }
}

// ---------------------------------------------------------------------------
// LSTM gate kernel: z (B x 4M, gate order i,f,g,o) -> c,h update.
// Writes fp32 h into d_out[:, t_step, :] and bf16 copies for next-step GEMMs.
// ---------------------------------------------------------------------------
__global__ __launch_bounds__(256)
void gate_kernel(const float* __restrict__ z,      // B x 4M
                 float* __restrict__ c,            // B x M
                 float* __restrict__ out,          // B x T x M
                 unsigned short* __restrict__ hc,  // B x 2M : [h ; c] bf16
                 unsigned short* __restrict__ az,  // B x (N+M) : h in cols [N, N+M)
                 int t_step)
{
    const int idx = blockIdx.x * blockDim.x + threadIdx.x;
    if (idx >= B_DIM * M_DIM) return;
    const int b = idx / M_DIM;
    const int m = idx - b * M_DIM;

    const float* zb = z + (size_t)b * 4 * M_DIM;
    const float zi = zb[m];
    const float zf = zb[M_DIM + m];
    const float zg = zb[2 * M_DIM + m];
    const float zo = zb[3 * M_DIM + m];

    const float cv = c[idx];
    const float cn = sigmoidf_(zf) * cv + sigmoidf_(zi) * htanh(zg);
    const float hn = sigmoidf_(zo) * htanh(cn);

    c[idx] = cn;
    out[((size_t)b * T_LEN + t_step) * M_DIM + m] = hn;
    hc[(size_t)b * 2 * M_DIM + m]          = f2bf(hn);
    hc[(size_t)b * 2 * M_DIM + M_DIM + m]  = f2bf(cn);
    az[(size_t)b * (N_DIM + M_DIM) + N_DIM + m] = f2bf(hn);
}

// ---------------------------------------------------------------------------
// Weight prep: dst[n*ld_dst + k_off + k] = bf16(src[k*Ncols + n])  (transpose)
// ---------------------------------------------------------------------------
__global__ __launch_bounds__(256)
void transpose_to_bf16(unsigned short* __restrict__ dst, int ld_dst, int k_off,
                       const float* __restrict__ src, int K, int Ncols)
{
    const int idx = blockIdx.x * blockDim.x + threadIdx.x;
    if (idx >= K * Ncols) return;
    const int k = idx / Ncols;
    const int n = idx - k * Ncols;
    dst[(size_t)n * ld_dst + k_off + k] = f2bf(src[(size_t)k * Ncols + n]);
}

// Xt[b,n,t] = bf16(X[b,t,n])  -- A matrix for the ux GEMM
__global__ __launch_bounds__(256)
void xt_kernel(unsigned short* __restrict__ Xt, const float* __restrict__ X)
{
    const int idx = blockIdx.x * blockDim.x + threadIdx.x;   // over B*N*T
    if (idx >= B_DIM * N_DIM * T_LEN) return;
    const int t = idx % T_LEN;
    const int r = idx / T_LEN;
    const int n = r % N_DIM;
    const int b = r / N_DIM;
    Xt[idx] = f2bf(X[((size_t)b * T_LEN + t) * N_DIM + n]);
}

// zero initial state: hc (B x 2M bf16), az (B x (N+M) bf16), c (B x M f32)
__global__ __launch_bounds__(256)
void init_state(unsigned short* __restrict__ hc,
                unsigned short* __restrict__ az,
                float* __restrict__ c)
{
    const int idx = blockIdx.x * blockDim.x + threadIdx.x;
    if (idx < B_DIM * 2 * M_DIM)       hc[idx] = 0;
    if (idx < B_DIM * (N_DIM + M_DIM)) az[idx] = 0;
    if (idx < B_DIM * M_DIM)           c[idx]  = 0.0f;
}

// ---------------------------------------------------------------------------
// Host-side orchestration
// ---------------------------------------------------------------------------
extern "C" void kernel_launch(void* const* d_in, const int* in_sizes, int n_in,
                              void* d_out, int out_size, void* d_ws, size_t ws_size,
                              hipStream_t stream)
{
    const float* X      = (const float*)d_in[0];  // B x T x N
    const float* We     = (const float*)d_in[1];  // 2M x T
    const float* be     = (const float*)d_in[2];  // T
    const float* Ue     = (const float*)d_in[3];  // T x T
    const float* bu     = (const float*)d_in[4];  // T
    const float* ve     = (const float*)d_in[5];  // T x 1
    const float* bv     = (const float*)d_in[6];  // 1
    const float* W_lstm = (const float*)d_in[7];  // N x 4M
    const float* U_lstm = (const float*)d_in[8];  // M x 4M
    const float* b_lstm = (const float*)d_in[9];  // 4M
    float* out = (float*)d_out;                   // B x T x M

    (void)in_sizes; (void)n_in; (void)out_size; (void)ws_size;

    // ---- carve workspace (256B aligned) ----
    char* ws = (char*)d_ws;
    size_t off = 0;
    auto carve = [&](size_t bytes) -> char* {
        char* p = ws + off;
        off = (off + bytes + 255) & ~(size_t)255;
        return p;
    };
    float*          ux  = (float*)         carve((size_t)B_DIM * N_DIM * T_LEN * 4); // 64 MB
    unsigned short* Xt  = (unsigned short*)carve((size_t)B_DIM * N_DIM * T_LEN * 2); // 32 MB
    unsigned short* WeT = (unsigned short*)carve((size_t)T_LEN * 2 * M_DIM * 2);
    unsigned short* UeT = (unsigned short*)carve((size_t)T_LEN * T_LEN * 2);
    unsigned short* WzT = (unsigned short*)carve((size_t)4 * M_DIM * (N_DIM + M_DIM) * 2);
    unsigned short* hc  = (unsigned short*)carve((size_t)B_DIM * 2 * M_DIM * 2);
    unsigned short* az  = (unsigned short*)carve((size_t)B_DIM * (N_DIM + M_DIM) * 2);
    float*          hs  = (float*)         carve((size_t)B_DIM * T_LEN * 4);
    float*          z   = (float*)         carve((size_t)B_DIM * 4 * M_DIM * 4);
    float*          c   = (float*)         carve((size_t)B_DIM * M_DIM * 4);

    // ---- one-time prep ----
    // WeT (T x 2M): transpose of We (2M x T)
    transpose_to_bf16<<<(2*M_DIM*T_LEN + 255)/256, 256, 0, stream>>>(
        WeT, 2*M_DIM, 0, We, 2*M_DIM, T_LEN);
    // UeT (T x T): transpose of Ue
    transpose_to_bf16<<<(T_LEN*T_LEN + 255)/256, 256, 0, stream>>>(
        UeT, T_LEN, 0, Ue, T_LEN, T_LEN);
    // WzT (4M x (N+M)) = transpose of [W_lstm ; U_lstm]
    transpose_to_bf16<<<(N_DIM*4*M_DIM + 255)/256, 256, 0, stream>>>(
        WzT, N_DIM + M_DIM, 0, W_lstm, N_DIM, 4*M_DIM);
    transpose_to_bf16<<<(M_DIM*4*M_DIM + 255)/256, 256, 0, stream>>>(
        WzT, N_DIM + M_DIM, N_DIM, U_lstm, M_DIM, 4*M_DIM);
    // Xt (B*N x T) = X transposed per batch
    xt_kernel<<<(B_DIM*N_DIM*T_LEN + 255)/256, 256, 0, stream>>>(Xt, X);
    // zero h0/c0 state
    init_state<<<(B_DIM*2*M_DIM + 255)/256, 256, 0, stream>>>(hc, az, c);

    // ux = Xt @ Ue + bu  :  (B*N x T) @ (T x T)
    {
        const int Mg = B_DIM * N_DIM, Ng = T_LEN, Kg = T_LEN;
        const int tiles = (Mg >> 5) * (Ng >> 5);
        gemm_bf16_wmma<<<(tiles + 7)/8, 256, 0, stream>>>(Xt, UeT, bu, ux, Mg, Ng, Kg);
    }

    // ---- sequential scan over T timesteps ----
    for (int t = 0; t < T_LEN; ++t) {
        // hs = [h;c] @ We + be : (B x 2M) @ (2M x T)
        {
            const int Mg = B_DIM, Ng = T_LEN, Kg = 2*M_DIM;
            const int tiles = (Mg >> 5) * (Ng >> 5);
            gemm_bf16_wmma<<<(tiles + 7)/8, 256, 0, stream>>>(hc, WeT, be, hs, Mg, Ng, Kg);
        }
        // attention + softmax + x_tilde
        attn_kernel<<<B_DIM, 256, 0, stream>>>(hs, ux, ve, bv, X, az, t);
        // z = [x_tilde ; h] @ [W_lstm ; U_lstm] + b_lstm : (B x 2M) @ (2M x 4M)
        {
            const int Mg = B_DIM, Ng = 4*M_DIM, Kg = N_DIM + M_DIM;
            const int tiles = (Mg >> 5) * (Ng >> 5);
            gemm_bf16_wmma<<<(tiles + 7)/8, 256, 0, stream>>>(az, WzT, b_lstm, z, Mg, Ng, Kg);
        }
        // gates -> c, h, outputs, next-step bf16 operands
        gate_kernel<<<(B_DIM*M_DIM + 255)/256, 256, 0, stream>>>(z, c, out, hc, az, t);
    }
}
